// TARGCN_47545287967480
// MI455X (gfx1250) — matrix operand order repaired and checked by reading, not compiled, not run on hardware
//
#include <hip/hip_runtime.h>
#include <math.h>

#define NUM_ENT 50000
#define NUM_REL 500
#define DIM 128
#define BQ 1024
#define KN 32
#define NPAD 50176           // 196 * 256, padded entity count for score tiling

typedef __attribute__((ext_vector_type(16))) __bf16 v16bf;
typedef __attribute__((ext_vector_type(8)))  float  v8f;

union BFr {
    v16bf v;
    __bf16 h[16];
    unsigned short u[16];
    unsigned int w32[8];
    uint4 q4[2];
};

// native conversion -> v_cvt_pk_bf16_f32 (RNE), instead of integer bit tricks
__device__ __forceinline__ unsigned short f2bf(float f) {
    __bf16 h = (__bf16)f;
    return __builtin_bit_cast(unsigned short, h);
}

__device__ __forceinline__ v8f wmma_bf16(v16bf a, v16bf b, v8f c) {
    return __builtin_amdgcn_wmma_f32_16x16x32_bf16(
        /*neg_a=*/false, a, /*neg_b=*/false, b,
        /*c_mod=*/(short)0, c, /*reuse_a=*/false, /*reuse_b=*/false);
}

// ---- operand loaders (wave32 WMMA VGPR layouts) ----
// B tile (32x16): lane l -> column N (caller folds in l%16); K in two runs of 8:
// elements 0..7 -> K = k0 + 8g + 0..7 ; elements 8..15 -> K = k0 + 16 + 8g + 0..7
__device__ __forceinline__ v16bf loadB_bf16(const unsigned short* __restrict__ W, int ldw,
                                            int n, int k0, int lane) {
    BFr r;
    const unsigned short* p = W + (size_t)n * ldw + k0 + ((lane >> 4) << 3);
    r.q4[0] = *(const uint4*)p;
    r.q4[1] = *(const uint4*)(p + 16);
    return r.v;
}

// A tile (16x32): lane l -> row M = row0 + l%16; K in two runs of 8:
// elements 0..7 -> K = k0 + 8g + 0..7 ; elements 8..15 -> K = k0 + 16 + 8g + 0..7
__device__ __forceinline__ v16bf loadA_lds(const unsigned short* __restrict__ S, int lds,
                                           int row0, int k0, int lane) {
    BFr r;
    const unsigned short* p = S + (size_t)(row0 + (lane & 15)) * lds + k0 + ((lane >> 4) << 3);
    r.q4[0] = *(const uint4*)p;
    r.q4[1] = *(const uint4*)(p + 16);
    return r.v;
}

__device__ __forceinline__ v16bf loadA_f32(const float* __restrict__ A, int lda,
                                           int row, int k0, int lane) {
    BFr r;
    const float* p = A + (size_t)row * lda + k0 + ((lane >> 4) << 3);
#pragma unroll
    for (int e = 0; e < 8; ++e) { r.h[e] = (__bf16)p[e]; r.h[8 + e] = (__bf16)p[16 + e]; }
    return r.v;
}

__device__ __forceinline__ v16bf loadA_bf16(const unsigned short* __restrict__ A, int lda,
                                            int row, int k0, int lane) {
    BFr r;
    const unsigned short* p = A + (size_t)row * lda + k0 + ((lane >> 4) << 3);
    r.q4[0] = *(const uint4*)p;
    r.q4[1] = *(const uint4*)(p + 16);
    return r.v;
}

// ---------------------------------------------------------------------------
// Kernel 0: one-shot f32 -> bf16 weight conversion (proj_w / gcn_w, 32K elems)
// ---------------------------------------------------------------------------
__global__ void __launch_bounds__(256) k_cvtw(const float* __restrict__ src,
                                              unsigned short* __restrict__ dst, int n) {
    int i = (blockIdx.x * 256 + threadIdx.x) * 4;
    if (i + 4 <= n) {
        float4 f = *(const float4*)(src + i);
        ushort4 o;
        o.x = f2bf(f.x); o.y = f2bf(f.y); o.z = f2bf(f.z); o.w = f2bf(f.w);
        *(ushort4*)(dst + i) = o;
    }
}

// ---------------------------------------------------------------------------
// Kernel 1: per-query neighbor encoder. One block per query b.
//   F[32][256] = [entity_emb | cos(dt*freq+phase)]   (bf16 in LDS)
//   node_emb = F @ proj_w^T + proj_b -> G[:, :128];  G[:,128:] = rel_emb
//   out = G @ gcn_w^T ; masked mean over K ; tanh ; * rel_q -> q[b, :]
// ---------------------------------------------------------------------------
#define FS 272   // LDS row stride (bf16 elems): 16B-aligned, conflict-skewed

__global__ void __launch_bounds__(256) k_subemb(
    const int* __restrict__ rel_idx, const int* __restrict__ ts,
    const int* __restrict__ ngh_node, const int* __restrict__ ngh_eidx,
    const int* __restrict__ ngh_ts,
    const float* __restrict__ symbol_emb,
    const unsigned short* __restrict__ gcn_wb,   // bf16 [128][256]
    const float* __restrict__ gcn_b,
    const unsigned short* __restrict__ proj_wb,  // bf16 [128][256]
    const float* __restrict__ proj_b,
    const float* __restrict__ basis_freq, const float* __restrict__ phase,
    unsigned short* __restrict__ q_out /* [BQ][DIM] bf16 */) {
    __shared__ unsigned short sF[KN][FS];
    __shared__ unsigned short sG[KN][FS];
    __shared__ float sAcc[DIM];
    __shared__ float sMask[KN];
    __shared__ int sCnt;

    const int b = blockIdx.x;
    const int tid = threadIdx.x;
    const int lane = tid & 31;
    const int w = tid >> 5;

    // ---- setup: build F, rel half of G, mask, count ----
    if (tid < DIM) sAcc[tid] = 0.f;
    if (tid < KN) sMask[tid] = (ngh_node[b * KN + tid] >= 0) ? 1.f : 0.f;
    if (tid == 0) {
        int c = 0;
        for (int k = 0; k < KN; ++k) c += (ngh_node[b * KN + k] >= 0) ? 1 : 0;
        sCnt = (c > 0) ? c : 1;
    }
    {
        int r = tid >> 3;              // neighbor row 0..31
        int c0 = (tid & 7) << 5;       // 32-col chunk
        int node = ngh_node[b * KN + r];
        int safe = (node >= 0) ? node : 0;
        float cut = (float)(ngh_ts[b * KN + r] - ts[b]);
        int eidx = ngh_eidx[b * KN + r] + NUM_ENT;
#pragma unroll
        for (int i = 0; i < 32; ++i) {
            int c = c0 + i;
            if (c < DIM) {
                sF[r][c] = f2bf(symbol_emb[(size_t)safe * DIM + c]);
            } else {
                int j = c - DIM;
                sF[r][c] = f2bf(cosf(cut * basis_freq[j] + phase[j]));
                sG[r][c] = f2bf(symbol_emb[(size_t)eidx * DIM + j]);
            }
        }
    }
    __syncthreads();

    // ---- GEMM1: node_emb = F @ proj_w^T + proj_b  -> sG[:, :128] ----
    {
        v8f c0v = {}, c1v = {};
        const int n0 = w << 4;
#pragma unroll
        for (int k0 = 0; k0 < 2 * DIM; k0 += 32) {
            v16bf bb = loadB_bf16(proj_wb, 2 * DIM, n0 + (lane & 15), k0, lane);
            v16bf a0 = loadA_lds(&sF[0][0], FS, 0, k0, lane);
            v16bf a1 = loadA_lds(&sF[0][0], FS, 16, k0, lane);
            c0v = wmma_bf16(a0, bb, c0v);
            c1v = wmma_bf16(a1, bb, c1v);
        }
        const int g = lane >> 4;
        const int col = n0 + (lane & 15);
        const float pb = proj_b[col];
#pragma unroll
        for (int v = 0; v < 8; ++v) {
            sG[(g << 3) + v][col]      = f2bf(c0v[v] + pb);
            sG[16 + (g << 3) + v][col] = f2bf(c1v[v] + pb);
        }
    }
    __syncthreads();

    // ---- GEMM2: out = G @ gcn_w^T ; masked sum over K via LDS atomics ----
    {
        v8f c0v = {}, c1v = {};
        const int n0 = w << 4;
#pragma unroll
        for (int k0 = 0; k0 < 2 * DIM; k0 += 32) {
            v16bf bb = loadB_bf16(gcn_wb, 2 * DIM, n0 + (lane & 15), k0, lane);
            v16bf a0 = loadA_lds(&sG[0][0], FS, 0, k0, lane);
            v16bf a1 = loadA_lds(&sG[0][0], FS, 16, k0, lane);
            c0v = wmma_bf16(a0, bb, c0v);
            c1v = wmma_bf16(a1, bb, c1v);
        }
        const int g = lane >> 4;
        const int col = n0 + (lane & 15);
        float s = 0.f;
#pragma unroll
        for (int v = 0; v < 8; ++v) {
            s += c0v[v] * sMask[(g << 3) + v];
            s += c1v[v] * sMask[16 + (g << 3) + v];
        }
        atomicAdd(&sAcc[col], s);   // ds_add_f32
    }
    __syncthreads();

    // ---- tanh(mean + bias) * rel_q -> q (bf16) ----
    if (tid < DIM) {
        float cnt = (float)sCnt;
        float sub = tanhf(sAcc[tid] / cnt + gcn_b[tid]);
        float rq = symbol_emb[(size_t)(NUM_ENT + rel_idx[b]) * DIM + tid];
        q_out[(size_t)b * DIM + tid] = f2bf(sub * rq);
    }
}

// ---------------------------------------------------------------------------
// Kernel 2a: bias2[d] = proj_b[d] + sum_j proj_w[d, 128+j] * cos(phase[j])
// (the constant time-encoding half of all_emb folds into this bias)
// ---------------------------------------------------------------------------
__global__ void __launch_bounds__(128) k_bias2(
    const float* __restrict__ proj_w, const float* __restrict__ proj_b,
    const float* __restrict__ phase, float* __restrict__ bias2) {
    int d = threadIdx.x;
    float s = proj_b[d];
    for (int j = 0; j < DIM; ++j)
        s += proj_w[(size_t)d * 2 * DIM + DIM + j] * cosf(phase[j]);
    bias2[d] = s;
}

// ---------------------------------------------------------------------------
// Kernel 2: all_emb[e,:] = symbol_emb[e,:128] @ proj_w[:, :128]^T + bias2
// 128 entity rows per block (one 16-row M-tile per wave), bf16 output,
// staged in LDS for coalesced 256B-row stores. Pad rows clamp (masked later).
// ---------------------------------------------------------------------------
__global__ void __launch_bounds__(256) k_allemb(
    const float* __restrict__ symbol_emb, const unsigned short* __restrict__ proj_wb,
    const float* __restrict__ bias2, unsigned short* __restrict__ allemb) {
    __shared__ unsigned short sOut[128][136];
    const int tid = threadIdx.x, lane = tid & 31, w = tid >> 5;
    const int g = lane >> 4;
    const int e0 = blockIdx.x * 128 + (w << 4);
    int erow = e0 + (lane & 15);
    if (erow >= NUM_ENT) erow = NUM_ENT - 1;   // pad rows: value irrelevant

    v16bf a[4];
#pragma unroll
    for (int kk = 0; kk < 4; ++kk)
        a[kk] = loadA_f32(symbol_emb, DIM, erow, kk * 32, lane);

#pragma unroll
    for (int nt = 0; nt < 8; ++nt) {
        v8f c = {};
        const int n = (nt << 4) + (lane & 15);
#pragma unroll
        for (int kk = 0; kk < 4; ++kk) {
            v16bf bb = loadB_bf16(proj_wb, 2 * DIM, n, kk * 32, lane);
            c = wmma_bf16(a[kk], bb, c);
        }
        const float bz = bias2[n];
#pragma unroll
        for (int v = 0; v < 8; ++v)
            sOut[(w << 4) + (g << 3) + v][n] = f2bf(c[v] + bz);
    }
    __syncthreads();

    // coalesced: 128 rows x 256B each
    {
        int r = tid >> 1, half = tid & 1;
        const uint4* src = (const uint4*)&sOut[r][half * 64];
        uint4* dst = (uint4*)(allemb + ((size_t)blockIdx.x * 128 + r) * DIM + half * 64);
#pragma unroll
        for (int i = 0; i < 8; ++i) dst[i] = src[i];
    }
}

// ---------------------------------------------------------------------------
// Kernel 3: score[b,e] = sum_d q[b,d] * all_emb[e,d]
// Block = 32 query rows x 256 entities; each wave: 2x2 WMMA C-tiles
// (A reused across 2 n-tiles, B reused across 2 m-tiles). f32 result staged
// in LDS, then streamed out as coalesced float4 (the 200MB HBM-bound part).
// ---------------------------------------------------------------------------
__global__ void __launch_bounds__(256) k_score(
    const unsigned short* __restrict__ q, const unsigned short* __restrict__ allemb,
    float* __restrict__ score) {
    __shared__ float sOut[32][264];
    const int tid = threadIdx.x, lane = tid & 31, w = tid >> 5;
    const int g = lane >> 4;
    const int b0 = blockIdx.y << 5;
    const int e0 = blockIdx.x << 8;
    const int nb = e0 + (w << 5);

    v8f c00 = {}, c01 = {}, c10 = {}, c11 = {};
#pragma unroll
    for (int kk = 0; kk < 4; ++kk) {
        const int k0 = kk << 5;
        v16bf a0 = loadA_bf16(q, DIM, b0 + (lane & 15), k0, lane);
        v16bf a1 = loadA_bf16(q, DIM, b0 + 16 + (lane & 15), k0, lane);
        v16bf bb0 = loadB_bf16(allemb, DIM, nb + (lane & 15), k0, lane);
        v16bf bb1 = loadB_bf16(allemb, DIM, nb + 16 + (lane & 15), k0, lane);
        c00 = wmma_bf16(a0, bb0, c00);
        c01 = wmma_bf16(a0, bb1, c01);
        c10 = wmma_bf16(a1, bb0, c10);
        c11 = wmma_bf16(a1, bb1, c11);
    }

    const int colb = (w << 5) + (lane & 15);
#pragma unroll
    for (int v = 0; v < 8; ++v) {
        const int r0 = (g << 3) + v;
        sOut[r0][colb]           = c00[v];
        sOut[r0][colb + 16]      = c01[v];
        sOut[r0 + 16][colb]      = c10[v];
        sOut[r0 + 16][colb + 16] = c11[v];
    }
    __syncthreads();

    {
        const int r = tid >> 3, seg = tid & 7;
        const float* src = &sOut[r][seg << 5];
        const size_t rowoff = (size_t)(b0 + r) * NUM_ENT;
        const int ebase = e0 + (seg << 5);
#pragma unroll
        for (int i = 0; i < 32; i += 4) {
            const int e = ebase + i;
            if (e + 4 <= NUM_ENT) {
                *(float4*)(score + rowoff + e) = *(const float4*)(src + i);
            } else {
                for (int j = 0; j < 4; ++j)
                    if (e + j < NUM_ENT) score[rowoff + e + j] = src[i + j];
            }
        }
    }
}

// ---------------------------------------------------------------------------
extern "C" void kernel_launch(void* const* d_in, const int* in_sizes, int n_in,
                              void* d_out, int out_size, void* d_ws, size_t ws_size,
                              hipStream_t stream) {
    // setup_inputs order: src_idx(unused), rel_idx, ts, ngh_node, ngh_eidx,
    // ngh_ts, symbol_emb, gcn_w_w, gcn_w_b, proj_w, proj_b, basis_freq, phase
    const int* rel_idx    = (const int*)d_in[1];
    const int* ts         = (const int*)d_in[2];
    const int* ngh_node   = (const int*)d_in[3];
    const int* ngh_eidx   = (const int*)d_in[4];
    const int* ngh_ts     = (const int*)d_in[5];
    const float* symbol_emb = (const float*)d_in[6];
    const float* gcn_w    = (const float*)d_in[7];
    const float* gcn_b    = (const float*)d_in[8];
    const float* proj_w   = (const float*)d_in[9];
    const float* proj_b   = (const float*)d_in[10];
    const float* basis    = (const float*)d_in[11];
    const float* phase    = (const float*)d_in[12];
    float* score = (float*)d_out;

    char* ws = (char*)d_ws;
    float* bias2 = (float*)ws;                                               // 512 B
    unsigned short* qbuf   = (unsigned short*)(ws + 4096);                   // 256 KB
    unsigned short* allemb = (unsigned short*)(ws + 4096 + (size_t)BQ * DIM * 2);          // 12.8 MB
    size_t aend = 4096 + (size_t)BQ * DIM * 2 + (size_t)NPAD * DIM * 2;
    unsigned short* proj_wb = (unsigned short*)(ws + aend);                  // 64 KB
    unsigned short* gcn_wb  = (unsigned short*)(ws + aend + DIM * 2 * DIM * 2);   // 64 KB

    const int wn = DIM * 2 * DIM;  // 32768 weight elements each
    k_cvtw<<<wn / (256 * 4), 256, 0, stream>>>(proj_w, proj_wb, wn);
    k_cvtw<<<wn / (256 * 4), 256, 0, stream>>>(gcn_w, gcn_wb, wn);
    k_bias2<<<1, 128, 0, stream>>>(proj_w, proj_b, phase, bias2);
    k_subemb<<<BQ, 256, 0, stream>>>(rel_idx, ts, ngh_node, ngh_eidx, ngh_ts,
                                     symbol_emb, gcn_wb, gcn_b, proj_wb, proj_b,
                                     basis, phase, qbuf);
    k_allemb<<<NPAD / 128, 256, 0, stream>>>(symbol_emb, proj_wb, bias2, allemb);
    k_score<<<dim3(NPAD / 256, BQ / 32), 256, 0, stream>>>(qbuf, allemb, score);
}
